// GriffinLim_64862596104660
// MI455X (gfx1250) — matrix operand order repaired and checked by reading, not compiled, not run on hardware
//
#include <hip/hip_runtime.h>
#include <hip/hip_bf16.h>
#include <math.h>

// ---------------------------------------------------------------------------
// Griffin-Lim on MI455X (gfx1250): STFT/iSTFT as WMMA f16 GEMMs against
// window-folded DFT bases.  All FFT work runs on v_wmma_f32_16x16x32_f16.
// K-loops are software-pipelined (double-buffered operands) so WMMAs issue
// back-to-back while the next K-step's tiles are in flight.
// ---------------------------------------------------------------------------

typedef __attribute__((ext_vector_type(16))) _Float16 v16h;
typedef __attribute__((ext_vector_type(8)))  _Float16 v8h;
typedef __attribute__((ext_vector_type(8)))  float    v8f;

#define B_SZ   32
#define T_SZ   1024
#define F_SZ   257
#define NFFT   512
#define HOP    160
#define PADW   256
#define FP     272                      // 257 padded up to multiple of 16
#define KC     (2*FP)                   // 544: [real | imag] combined K dim
#define OUTLEN (NFFT + HOP*(T_SZ-1))    // 164192 (center-padded length)
#define WLEN   (OUTLEN - 2*PADW)        // 163680 (trimmed waveform length)
#define N_ITER 32
#define TWO_PI_F 6.28318530717958647692f

__device__ __forceinline__ void load16(const _Float16* p, v16h& dst) {
    *(v8h*)&dst       = *(const v8h*)(p);
    *((v8h*)&dst + 1) = *(const v8h*)(p + 8);
}
__device__ __forceinline__ void loadA16(const _Float16* p, v16h& dst) {
    // A K-layout: 16 contiguous K (lane offset pre-applied by caller)
    *(v8h*)&dst       = *(const v8h*)(p);
    *((v8h*)&dst + 1) = *(const v8h*)(p + 8);
}

// ---------------------------------------------------------------------------
// Basis matrices (f16, stored transposed so WMMA B-lanes read contiguous K).
//   BfwdT[j][n]  j in [0,544): j<272 -> w[n]*cos(2*pi*j*n/512)
//                              j>=272 -> -w[n]*sin(2*pi*(j-272)*n/512)
//   BinvT[n][k]  k<272 -> c_k*cos(..)*w[n]/512 ; k>=272 -> -c_k*sin(..)*w[n]/512
//   with c_0=c_256=1, else 2 (irfft), pad bins (kk>256) = 0.
// ---------------------------------------------------------------------------
__global__ void build_mats(const float* __restrict__ win,
                           _Float16* __restrict__ BinvT,
                           _Float16* __restrict__ BfwdT) {
    int idx  = blockIdx.x * blockDim.x + threadIdx.x;
    int half = KC * NFFT;
    if (idx >= 2 * half) return;
    if (idx >= half) {
        int i  = idx - half;
        int j  = i / NFFT;              // output column (freq, re/im)
        int n  = i % NFFT;              // time sample
        int kk = (j < FP) ? j : j - FP;
        float v = 0.f;
        if (kk <= NFFT / 2) {
            float ph = TWO_PI_F * (float)kk * (float)n / (float)NFFT;
            v = (j < FP) ? win[n] * __cosf(ph) : -win[n] * __sinf(ph);
        }
        BfwdT[(size_t)j * NFFT + n] = (_Float16)v;
    } else {
        int i  = idx;
        int n  = i / KC;                // time sample
        int k  = i % KC;                // combined spec index
        int kk = (k < FP) ? k : k - FP;
        float v = 0.f;
        if (kk <= NFFT / 2) {
            float c  = (kk == 0 || kk == NFFT / 2) ? 1.f : 2.f;
            float ph = TWO_PI_F * (float)kk * (float)n / (float)NFFT;
            float s  = c * win[n] * (1.f / (float)NFFT);
            v = (k < FP) ? s * __cosf(ph) : -s * __sinf(ph);
        }
        BinvT[(size_t)n * KC + k] = (_Float16)v;
    }
}

// NOLA normalization lookup: inv_wsq[pos] over the padded coordinate system.
__global__ void build_wsq(const float* __restrict__ win,
                          float* __restrict__ inv_wsq) {
    int pos = blockIdx.x * blockDim.x + threadIdx.x;
    if (pos >= OUTLEN) return;
    int tlo = (pos - (NFFT - 1) + (HOP - 1)) / HOP; if (tlo < 0) tlo = 0;
    int thi = pos / HOP;                            if (thi > T_SZ - 1) thi = T_SZ - 1;
    float s = 0.f;
    for (int t = tlo; t <= thi; ++t) { float w = win[pos - t * HOP]; s += w * w; }
    inv_wsq[pos] = (s > 1e-11f) ? 1.f / s : 1.f;
}

// spec = mag * exp(i*theta) -> f16 A-matrix row [re(0..271) | im(0..271)]
__global__ void build_spec(const float* __restrict__ mag,
                           const float* __restrict__ ang,
                           _Float16* __restrict__ specA) {
    int idx = blockIdx.x * blockDim.x + threadIdx.x;
    if (idx >= B_SZ * T_SZ * FP) return;
    int f  = idx % FP;
    int bt = idx / FP;
    float re = 0.f, im = 0.f;
    if (f < F_SZ) {
        float m = mag[(size_t)bt * F_SZ + f];
        float a = ang[(size_t)bt * F_SZ + f];
        float s, c;
        __sincosf(a, &s, &c);
        re = m * c; im = m * s;
    }
    specA[(size_t)bt * KC + f]      = (_Float16)re;
    specA[(size_t)bt * KC + FP + f] = (_Float16)im;
}

// ---------------------------------------------------------------------------
// iSTFT GEMM: frames[B*T, 512] = specA[B*T, 544] x Binv[544, 512]
// One wave per 16x64 output tile: 4 accumulators share one A load per K-step.
// Double-buffered operands: the loop loads step s+1 while WMMAing step s.
// ---------------------------------------------------------------------------
__global__ void __launch_bounds__(256)
istft_mm(const _Float16* __restrict__ specA,
         const _Float16* __restrict__ BinvT,
         float* __restrict__ frames) {
    const int lane  = threadIdx.x & 31;
    const int wave  = blockIdx.x * (blockDim.x >> 5) + (threadIdx.x >> 5);
    const int g     = wave & 7;         // 8 groups of 4 N-tiles over NFFT
    const int mtile = wave >> 3;        // 2048 tiles over B*T
    const int l15   = lane & 15;
    const int hi    = lane >> 4;        // which lane-half
    const int aoff0 = hi * 16;          // A K-layout: {0..15} / {16..31}
    const int boff  = hi * 16;          // B lanes: 16 contiguous K per half

    const _Float16* arow  = specA + (size_t)(mtile * 16 + l15) * KC + aoff0;
    const _Float16* brow0 = BinvT + (size_t)(g * 64 + 0 * 16 + l15) * KC + boff;
    const _Float16* brow1 = BinvT + (size_t)(g * 64 + 1 * 16 + l15) * KC + boff;
    const _Float16* brow2 = BinvT + (size_t)(g * 64 + 2 * 16 + l15) * KC + boff;
    const _Float16* brow3 = BinvT + (size_t)(g * 64 + 3 * 16 + l15) * KC + boff;

    v8f acc0 = {}, acc1 = {}, acc2 = {}, acc3 = {};
    v16h a, b0, b1, b2, b3;
    loadA16(arow, a);
    load16(brow0, b0); load16(brow1, b1); load16(brow2, b2); load16(brow3, b3);

    #pragma unroll 2
    for (int s = 0; s < KC / 32 - 1; ++s) {
        const int k1 = (s + 1) * 32;
        v16h an, b0n, b1n, b2n, b3n;
        loadA16(arow + k1, an);
        load16(brow0 + k1, b0n); load16(brow1 + k1, b1n);
        load16(brow2 + k1, b2n); load16(brow3 + k1, b3n);
        acc0 = __builtin_amdgcn_wmma_f32_16x16x32_f16(false, a, false, b0,
                                                      (short)0, acc0, false, false);
        acc1 = __builtin_amdgcn_wmma_f32_16x16x32_f16(false, a, false, b1,
                                                      (short)0, acc1, false, false);
        acc2 = __builtin_amdgcn_wmma_f32_16x16x32_f16(false, a, false, b2,
                                                      (short)0, acc2, false, false);
        acc3 = __builtin_amdgcn_wmma_f32_16x16x32_f16(false, a, false, b3,
                                                      (short)0, acc3, false, false);
        a = an; b0 = b0n; b1 = b1n; b2 = b2n; b3 = b3n;
    }
    acc0 = __builtin_amdgcn_wmma_f32_16x16x32_f16(false, a, false, b0,
                                                  (short)0, acc0, false, false);
    acc1 = __builtin_amdgcn_wmma_f32_16x16x32_f16(false, a, false, b1,
                                                  (short)0, acc1, false, false);
    acc2 = __builtin_amdgcn_wmma_f32_16x16x32_f16(false, a, false, b2,
                                                  (short)0, acc2, false, false);
    acc3 = __builtin_amdgcn_wmma_f32_16x16x32_f16(false, a, false, b3,
                                                  (short)0, acc3, false, false);

    // C layout: VGPR r, lane-half hi -> (M = hi*8 + r, N = l15)
    float* out = frames + (size_t)(mtile * 16 + hi * 8) * NFFT + g * 64 + l15;
    #pragma unroll
    for (int r = 0; r < 8; ++r) {
        out[(size_t)r * NFFT + 0]  = acc0[r];
        out[(size_t)r * NFFT + 16] = acc1[r];
        out[(size_t)r * NFFT + 32] = acc2[r];
        out[(size_t)r * NFFT + 48] = acc3[r];
    }
}

// Overlap-add + NOLA normalize (gather form: <=4 frames per sample, no atomics).
__global__ void ola(const float* __restrict__ frames,
                    const float* __restrict__ inv_wsq,
                    float* __restrict__ out) {
    int idx = blockIdx.x * blockDim.x + threadIdx.x;
    if (idx >= B_SZ * WLEN) return;
    int b   = idx / WLEN;
    int m   = idx % WLEN;
    int pos = m + PADW;
    int tlo = (pos - (NFFT - 1) + (HOP - 1)) / HOP; if (tlo < 0) tlo = 0;
    int thi = pos / HOP;                            if (thi > T_SZ - 1) thi = T_SZ - 1;
    const float* fb = frames + (size_t)b * T_SZ * NFFT;
    float s = 0.f;
    for (int t = tlo; t <= thi; ++t) s += fb[(size_t)t * NFFT + (pos - t * HOP)];
    out[idx] = s * inv_wsq[pos];
}

// Re-frame the waveform for the forward STFT: f16 frames with virtual reflect
// padding.  Each thread writes one 16-byte v8h chunk.
__global__ void build_frames(const float* __restrict__ ytrim,
                             _Float16* __restrict__ framesF16) {
    int idx = blockIdx.x * blockDim.x + threadIdx.x;
    if (idx >= B_SZ * T_SZ * (NFFT / 8)) return;
    int chunk = idx & 63;               // which 8-sample chunk of the frame
    int row   = idx >> 6;               // b*T + t
    int bb    = row >> 10;
    int tt    = row & (T_SZ - 1);
    const float* ysrc = ytrim + (size_t)bb * WLEN;
    v8h v;
    #pragma unroll
    for (int i = 0; i < 8; ++i) {
        int pos = tt * HOP + chunk * 8 + i;
        int r   = pos - PADW;           // reflect into [0, WLEN)
        r = (r < 0) ? -r : r;
        if (r >= WLEN) r = 2 * WLEN - 2 - r;
        v[i] = (_Float16)ysrc[r];
    }
    *(v8h*)(framesF16 + (size_t)row * NFFT + chunk * 8) = v;
}

// ---------------------------------------------------------------------------
// STFT GEMM + phase: rec[B*T, 544] = framesF16[B*T, 512] x Bfwd[512, 544]
// One wave per (16-row, 16-freq) tile pair; real+imag accumulators,
// 2 WMMA per K-step, double-buffered operands, then atan2.
// ---------------------------------------------------------------------------
__global__ void __launch_bounds__(256)
stft_mm_angles(const _Float16* __restrict__ framesF16,
               const _Float16* __restrict__ BfwdT,
               float* __restrict__ angles) {
    const int lane  = threadIdx.x & 31;
    const int wave  = blockIdx.x * (blockDim.x >> 5) + (threadIdx.x >> 5);
    const int jtile = wave % (FP / 16);     // 17 freq tiles
    const int mtile = wave / (FP / 16);     // 2048 row tiles
    const int l15   = lane & 15;
    const int hi    = lane >> 4;
    const int aoff0 = hi * 16;
    const int boff  = hi * 16;

    const _Float16* arow = framesF16 + (size_t)(mtile * 16 + l15) * NFFT + aoff0;
    const _Float16* brr  = BfwdT + (size_t)(jtile * 16 + l15) * NFFT + boff;       // real
    const _Float16* bri  = BfwdT + (size_t)(FP + jtile * 16 + l15) * NFFT + boff;  // imag

    v8f accR = {}, accI = {};
    v16h a, br, bi;
    loadA16(arow, a);
    load16(brr, br); load16(bri, bi);

    #pragma unroll 2
    for (int s = 0; s < NFFT / 32 - 1; ++s) {
        const int k1 = (s + 1) * 32;
        v16h an, brn, bin;
        loadA16(arow + k1, an);
        load16(brr + k1, brn); load16(bri + k1, bin);
        accR = __builtin_amdgcn_wmma_f32_16x16x32_f16(false, a, false, br,
                                                      (short)0, accR, false, false);
        accI = __builtin_amdgcn_wmma_f32_16x16x32_f16(false, a, false, bi,
                                                      (short)0, accI, false, false);
        a = an; br = brn; bi = bin;
    }
    accR = __builtin_amdgcn_wmma_f32_16x16x32_f16(false, a, false, br,
                                                  (short)0, accR, false, false);
    accI = __builtin_amdgcn_wmma_f32_16x16x32_f16(false, a, false, bi,
                                                  (short)0, accI, false, false);

    const int fcol = jtile * 16 + l15;
    if (fcol < F_SZ) {
        float* aout = angles + (size_t)(mtile * 16 + hi * 8) * F_SZ + fcol;
        #pragma unroll
        for (int r = 0; r < 8; ++r) aout[(size_t)r * F_SZ] = atan2f(accI[r], accR[r]);
    }
}

// ---------------------------------------------------------------------------
// Host-side orchestration
// ---------------------------------------------------------------------------
extern "C" void kernel_launch(void* const* d_in, const int* in_sizes, int n_in,
                              void* d_out, int out_size, void* d_ws, size_t ws_size,
                              hipStream_t stream) {
    const float* mag  = (const float*)d_in[0];   // [32,1024,257]
    const float* ang0 = (const float*)d_in[1];   // [32,1024,257]
    const float* win  = (const float*)d_in[2];   // [512]
    float* out = (float*)d_out;                  // [32,163680]

    char* ws = (char*)d_ws;
    size_t off = 0;
    auto alloc = [&](size_t bytes) -> void* {
        void* p = ws + off;
        off = (off + bytes + 255) & ~(size_t)255;
        return p;
    };
    _Float16* BinvT  = (_Float16*)alloc(sizeof(_Float16) * (size_t)NFFT * KC);
    _Float16* BfwdT  = (_Float16*)alloc(sizeof(_Float16) * (size_t)KC * NFFT);
    float*    invw   = (float*)   alloc(sizeof(float)    * (size_t)OUTLEN);
    float*    angles = (float*)   alloc(sizeof(float)    * (size_t)B_SZ * T_SZ * F_SZ);
    _Float16* specA  = (_Float16*)alloc(sizeof(_Float16) * (size_t)B_SZ * T_SZ * KC);
    float*    frames = (float*)   alloc(sizeof(float)    * (size_t)B_SZ * T_SZ * NFFT);
    _Float16* framesH= (_Float16*)alloc(sizeof(_Float16) * (size_t)B_SZ * T_SZ * NFFT);
    float*    ytrim  = (float*)   alloc(sizeof(float)    * (size_t)B_SZ * WLEN);
    (void)ws_size; (void)in_sizes; (void)n_in; (void)out_size;

    const int nMats = 2 * KC * NFFT;
    build_mats<<<(nMats + 255) / 256, 256, 0, stream>>>(win, BinvT, BfwdT);
    build_wsq<<<(OUTLEN + 255) / 256, 256, 0, stream>>>(win, invw);

    const int specN   = B_SZ * T_SZ * FP;
    const int frameN  = B_SZ * T_SZ * (NFFT / 8);
    const int istftBl = (B_SZ * T_SZ / 16) * (NFFT / 64) / 8;   // 2048 blocks (8 waves ea)
    const int stftBl  = (B_SZ * T_SZ / 16) * (FP / 16) / 8;     // 4352 blocks
    const int olaN    = B_SZ * WLEN;

    for (int it = 0; it < N_ITER; ++it) {
        const float* asrc = (it == 0) ? ang0 : angles;
        build_spec<<<(specN + 255) / 256, 256, 0, stream>>>(mag, asrc, specA);
        istft_mm<<<istftBl, 256, 0, stream>>>(specA, BinvT, frames);
        ola<<<(olaN + 255) / 256, 256, 0, stream>>>(frames, invw, ytrim);
        build_frames<<<(frameN + 255) / 256, 256, 0, stream>>>(ytrim, framesH);
        stft_mm_angles<<<stftBl, 256, 0, stream>>>(framesH, BfwdT, angles);
    }

    // Final synthesis with the converged phases -> d_out
    build_spec<<<(specN + 255) / 256, 256, 0, stream>>>(mag, angles, specA);
    istft_mm<<<istftBl, 256, 0, stream>>>(specA, BinvT, frames);
    ola<<<(olaN + 255) / 256, 256, 0, stream>>>(frames, invw, out);
}
